// SimpleNet_73409581023694
// MI455X (gfx1250) — matrix operand order repaired and checked by reading, not compiled, not run on hardware
//
#include <hip/hip_runtime.h>
#include <hip/hip_bf16.h>
#include <cstdint>

#define BS 32
#define T  1024
#define J  64
#define D  256
#define G3 768          // 3*D
#define D5 1280         // 5*D
#define NL 32           // GRU scan length (reference scans over BS axis)

typedef __attribute__((ext_vector_type(16))) __bf16 v16bf;
typedef __attribute__((ext_vector_type(8)))  float  v8f;

__device__ __forceinline__ unsigned short f2bf(float f) {
  unsigned x = __float_as_uint(f);
  x += 0x7FFFu + ((x >> 16) & 1u);       // round-to-nearest-even
  return (unsigned short)(x >> 16);
}
__device__ __forceinline__ float sigmoidf(float x) { return 1.0f / (1.0f + expf(-x)); }

// ---------------------------------------------------------------------------
// gfx1250 async global->LDS staging (ASYNCcnt path), with sync fallback.
// Builtin signature (from compiler diagnostic): pointers are int4-vector
// typed: (int4 AS1* src, int4 AS3* dst, const int offset, const int cpol).
// ---------------------------------------------------------------------------
#if __has_builtin(__builtin_amdgcn_global_load_async_to_lds_b128) && \
    __has_builtin(__builtin_amdgcn_s_wait_asynccnt)
#define USE_ASYNC_LDS 1
#else
#define USE_ASYNC_LDS 0
#endif

#if USE_ASYNC_LDS
typedef int v4i_vs __attribute__((vector_size(16)));

__device__ __forceinline__ __attribute__((address_space(1))) v4i_vs* to_glb(const void* p) {
  return (__attribute__((address_space(1))) v4i_vs*)(uintptr_t)p;
}
// Generic LDS pointers on CDNA5 carry the LDS byte offset in addr[31:0]
// (ISA 10.2 aperture rules), so truncation gives a valid AS(3) pointer.
__device__ __forceinline__ __attribute__((address_space(3))) v4i_vs* to_lds(void* p) {
  return (__attribute__((address_space(3))) v4i_vs*)(uintptr_t)(unsigned)(uintptr_t)p;
}
#endif

// ---------------------------------------------------------------------------
// Generic tiled bf16 GEMM:  C[M,N](f32) = A[M,K](bf16) * B[N,K](bf16)^T + bias
// Both operands K-major.  64x64 tile per 256-thread block, BK=32 per WMMA,
// double-buffered LDS staging. Batched via blockIdx.z (strides sA/sB/sC).
// OOB tile rows/cols are index-clamped: A row r only feeds C row r (B col c
// only feeds C col c), and those C entries are never stored, so clamped
// duplicates are harmless and the staging path stays branch-free.
// ---------------------------------------------------------------------------
#define BM 64
#define BN 64
#define BK 32
#define LDP 20   // LDS row stride in dwords: 80B rows -> 16B-aligned lane
                 // stores for async b128, and order-16 mod 64 banks.

__global__ __launch_bounds__(256) void gemm_bf16(
    const unsigned short* __restrict__ A,
    const unsigned short* __restrict__ B,
    float* __restrict__ C,
    const float* __restrict__ bias,
    int M, int N, int K,
    long sA, long sB, long sC)
{
  __shared__ unsigned As[2][BM][LDP];
  __shared__ unsigned Bs[2][BN][LDP];
  A += (long)blockIdx.z * sA;
  B += (long)blockIdx.z * sB;
  C += (long)blockIdx.z * sC;
  const int m0   = blockIdx.x * BM;
  const int n0   = blockIdx.y * BN;
  const int tid  = threadIdx.x;
  const int lane = tid & 31;
  const int wave = tid >> 5;          // 0..7
  const int lrow = tid >> 2;          // 0..63  (tile row for staging)
  const int lk   = (tid & 3) << 3;    // 0,8,16,24 (k offset, 8 bf16 = 16B)
  const int kp   = lk >> 1;           // dword index within LDS row
  const int cn   = wave & 3;          // N-subtile of this wave
  const int rm   = (wave >> 2) << 1;  // M-subtile base: 0 or 2
  const int mrow = lane & 15;
  const int hi   = lane >> 4;         // 0/1 half-wave

  int gr = m0 + lrow; if (gr > M - 1) gr = M - 1;   // clamp (see note above)
  int gc = n0 + lrow; if (gc > N - 1) gc = N - 1;
  const unsigned short* ga = A + (long)gr * K + lk;
  const unsigned short* gb = B + (long)gc * K + lk;

  v8f acc0 = {0.f,0.f,0.f,0.f,0.f,0.f,0.f,0.f};
  v8f acc1 = {0.f,0.f,0.f,0.f,0.f,0.f,0.f,0.f};

  const int nk = K / BK;

  auto stage = [&](int i) {
    const int b = i & 1;
#if USE_ASYNC_LDS
    __builtin_amdgcn_global_load_async_to_lds_b128(
        to_glb(ga + (long)i * BK), to_lds(&As[b][lrow][kp]), 0, 0);
    __builtin_amdgcn_global_load_async_to_lds_b128(
        to_glb(gb + (long)i * BK), to_lds(&Bs[b][lrow][kp]), 0, 0);
#else
    const uint4 va = *(const uint4*)(ga + (long)i * BK);
    const uint4 vb = *(const uint4*)(gb + (long)i * BK);
    As[b][lrow][kp] = va.x; As[b][lrow][kp+1] = va.y;
    As[b][lrow][kp+2] = va.z; As[b][lrow][kp+3] = va.w;
    Bs[b][lrow][kp] = vb.x; Bs[b][lrow][kp+1] = vb.y;
    Bs[b][lrow][kp+2] = vb.z; Bs[b][lrow][kp+3] = vb.w;
#endif
  };

  stage(0);
  for (int i = 0; i < nk; ++i) {
    const int cb = i & 1;
#if USE_ASYNC_LDS
    __builtin_amdgcn_s_wait_asynccnt(0);   // my staging of buffer cb is done
#endif
    __syncthreads();                       // everyone's staging done; previous
                                           // compute on buffer 1-cb finished
    if (i + 1 < nk) stage(i + 1);          // prefetch next chunk (other buffer)

    // Build WMMA fragments per CDNA5 16-bit A (16x32) / B (32x16) layouts.
    union Frag { unsigned u[8]; v16bf v; } a0, a1, b;
    #pragma unroll
    for (int vi = 0; vi < 8; ++vi) {
      // A: k = (vi<4?0:16) + (hi?8:0) + (vi&3)*2 -> pair index:
      const int kpa = ((vi & 4) << 1) + (hi << 2) + (vi & 3);
      a0.u[vi] = As[cb][rm * 16 + mrow][kpa];
      a1.u[vi] = As[cb][rm * 16 + 16 + mrow][kpa];
      // B: same k as A -> pair index kpa ; col = cn*16+mrow
      b.u[vi]  = Bs[cb][cn * 16 + mrow][kpa];
    }
    acc0 = __builtin_amdgcn_wmma_f32_16x16x32_bf16(false, a0.v, false, b.v,
                                                   (short)0, acc0, false, false);
    acc1 = __builtin_amdgcn_wmma_f32_16x16x32_bf16(false, a1.v, false, b.v,
                                                   (short)0, acc1, false, false);
  }

  // C/D layout: lane 0-15 -> N=lane, M=i ; lane 16-31 -> N=lane-16, M=8+i
  const int ncol = n0 + cn * 16 + mrow;
  if (ncol < N) {
    const float bv = bias ? bias[ncol] : 0.0f;
    #pragma unroll
    for (int i = 0; i < 8; ++i) {
      const int r0 = m0 + rm * 16 + (hi ? 8 : 0) + i;
      if (r0 < M)      C[(long)r0 * N + ncol]        = acc0[i] + bv;
      if (r0 + 16 < M) C[(long)(r0 + 16) * N + ncol] = acc1[i] + bv;
    }
  }
}

// ---------------------------------------------------------------------------
// Helpers
// ---------------------------------------------------------------------------
__global__ void zero_ws(float* p, long n) {
  for (long i = (long)blockIdx.x * 256 + threadIdx.x; i < n; i += (long)gridDim.x * 256)
    p[i] = 0.0f;
}

__global__ void cvt_bf(const float* __restrict__ in, unsigned short* __restrict__ out, long n) {
  for (long i = (long)blockIdx.x * 256 + threadIdx.x; i < n; i += (long)gridDim.x * 256)
    out[i] = f2bf(in[i]);
}

// ec gather: one block per (bs,t) row; also builds bf16 ec*w3 for the s2 GEMM.
// (s1 = ec@(w1+w2) is dropped: softmax is shift-invariant along j.)
__global__ __launch_bounds__(256) void gather_ctx(
    const int* __restrict__ ctx, const float* __restrict__ emb,
    const float* __restrict__ watt, float* __restrict__ ec,
    unsigned short* __restrict__ ecw3)
{
  const long row = blockIdx.x;
  const int  d   = threadIdx.x;
  const int  tok = ctx[row];
  const float e  = emb[(long)tok * D + d];
  ec[row * D + d]   = e;
  ecw3[row * D + d] = f2bf(e * watt[2 * D + d]);
}

// eq gather: bf16 eq (J x D, K-major for s2's B operand) and bf16 eq^T (D x J,
// K-major for c2q's B operand).
__global__ __launch_bounds__(256) void gather_q(
    const int* __restrict__ qry, const float* __restrict__ emb,
    unsigned short* __restrict__ eqb, unsigned short* __restrict__ eqT)
{
  const int row = blockIdx.x;           // bs*J + j
  const int d   = threadIdx.x;
  const int bs  = row >> 6, j = row & 63;
  const int tok = qry[row];
  const float e = emb[(long)tok * D + d];
  const unsigned short b = f2bf(e);
  eqb[(long)row * D + d] = b;
  eqT[((long)bs * D + d) * J + j] = b;
}

// Row softmax over j=64; 4 rows per 256-thread block; emits bf16 A matrix.
__global__ __launch_bounds__(256) void softmax_att(
    const float* __restrict__ s2, unsigned short* __restrict__ att)
{
  const int r = threadIdx.x >> 6;
  const int j = threadIdx.x & 63;
  const long row = (long)blockIdx.x * 4 + r;
  __shared__ float red[4][64];
  const float x = s2[row * J + j];
  red[r][j] = x; __syncthreads();
  for (int s = 32; s > 0; s >>= 1) { if (j < s) red[r][j] = fmaxf(red[r][j], red[r][j + s]); __syncthreads(); }
  const float mx = red[r][0]; __syncthreads();
  const float ex = expf(x - mx);
  red[r][j] = ex; __syncthreads();
  for (int s = 32; s > 0; s >>= 1) { if (j < s) red[r][j] += red[r][j + s]; __syncthreads(); }
  att[row * J + j] = f2bf(ex / red[r][0]);
}

// G = [ec, c2q, ec*c2q] in bf16 (feeds the two big xW GEMMs).
__global__ __launch_bounds__(256) void build_g(
    const float* __restrict__ ec, const float* __restrict__ c2q,
    unsigned short* __restrict__ G)
{
  const long row = blockIdx.x;
  const int  d   = threadIdx.x;
  const float e = ec[row * D + d];
  const float c = c2q[row * D + d];
  unsigned short* g = G + row * G3;
  g[d]         = f2bf(e);
  g[D + d]     = f2bf(c);
  g[2 * D + d] = f2bf(e * c);
}

// sum over t of G[bs,t,:] (f32, recomputed from ec/c2q for accuracy).
__global__ __launch_bounds__(256) void sum_g(
    const float* __restrict__ ec, const float* __restrict__ c2q,
    float* __restrict__ sumG)
{
  const int bs  = blockIdx.x / 12;
  const int cg  = blockIdx.x % 12;
  const int col = cg * 64 + (threadIdx.x & 63);
  const int tl  = threadIdx.x >> 6;     // 0..3
  float acc = 0.0f;
  for (int t = tl; t < T; t += 4) {
    const long r = ((long)bs * T + t) * D;
    float v;
    if (col < D)          v = ec[r + col];
    else if (col < 2 * D) v = c2q[r + col - D];
    else                  v = ec[r + col - 2 * D] * c2q[r + col - 2 * D];
    acc += v;
  }
  __shared__ float red[256];
  red[threadIdx.x] = acc;
  __syncthreads();
  if (tl == 0)
    sumG[(long)bs * G3 + col] =
        red[threadIdx.x] + red[threadIdx.x + 64] + red[threadIdx.x + 128] + red[threadIdx.x + 192];
}

// GRU gate fusion for one scan step: consumes xW slice (has bih) and hW (has
// bhh), updates h (f32 + bf16 for the next recurrent GEMM), accumulates the
// per-step column sum that becomes GM's Mf/Mb part.
__global__ __launch_bounds__(256) void gru_gate(
    const float* __restrict__ xW,     // (T x G3), already offset to step l
    const float* __restrict__ hW,     // (T x G3)
    float* __restrict__ h,            // (T x D)
    unsigned short* __restrict__ h_bf,
    float* __restrict__ sumM)         // (D), already offset to step l
{
  const int c = threadIdx.x;          // 0..255
  float local = 0.0f;
  for (int tt = 0; tt < 64; ++tt) {
    const int t = blockIdx.x * 64 + tt;
    const float* xw = xW + (long)t * G3;
    const float* hw = hW + (long)t * G3;
    const float r = sigmoidf(xw[c] + hw[c]);
    const float z = sigmoidf(xw[D + c] + hw[D + c]);
    const float n = tanhf(xw[2 * D + c] + r * hw[2 * D + c]);
    const float ho = h[(long)t * D + c];
    const float hn = (1.0f - z) * n + z * ho;
    h[(long)t * D + c]    = hn;
    h_bf[(long)t * D + c] = f2bf(hn);
    local += hn;
  }
  atomicAdd(&sumM[c], local);
}

// GM = [sumG | sumMf | sumMb] -> bf16 for the final logits GEMM.
__global__ __launch_bounds__(256) void build_gm(
    const float* __restrict__ sumG, const float* __restrict__ sumMf,
    const float* __restrict__ sumMb, unsigned short* __restrict__ GM)
{
  const int bs = blockIdx.x;
  for (int k = threadIdx.x; k < D5; k += 256) {
    float v;
    if (k < G3)          v = sumG[(long)bs * G3 + k];
    else if (k < G3 + D) v = sumMf[(long)bs * D + (k - G3)];
    else                 v = sumMb[(long)bs * D + (k - G3 - D)];
    GM[(long)bs * D5 + k] = f2bf(v);
  }
}

__global__ __launch_bounds__(256) void logsoftmax_rows(
    const float* __restrict__ logits, float* __restrict__ out)
{
  const int row = blockIdx.x;
  const int tid = threadIdx.x;
  __shared__ float red[256];
  float mx = -3.4e38f;
  for (int i = tid; i < T; i += 256) mx = fmaxf(mx, logits[(long)row * T + i]);
  red[tid] = mx; __syncthreads();
  for (int s = 128; s > 0; s >>= 1) { if (tid < s) red[tid] = fmaxf(red[tid], red[tid + s]); __syncthreads(); }
  const float M = red[0]; __syncthreads();
  float sm = 0.0f;
  for (int i = tid; i < T; i += 256) sm += expf(logits[(long)row * T + i] - M);
  red[tid] = sm; __syncthreads();
  for (int s = 128; s > 0; s >>= 1) { if (tid < s) red[tid] += red[tid + s]; __syncthreads(); }
  const float lse = M + logf(red[0]);
  for (int i = tid; i < T; i += 256) out[(long)row * T + i] = logits[(long)row * T + i] - lse;
}

// ---------------------------------------------------------------------------
extern "C" void kernel_launch(void* const* d_in, const int* in_sizes, int n_in,
                              void* d_out, int out_size, void* d_ws, size_t ws_size,
                              hipStream_t stream) {
  (void)in_sizes; (void)n_in; (void)out_size; (void)ws_size;
  const int*   ctx   = (const int*)d_in[0];
  const int*   qry   = (const int*)d_in[1];
  const float* emb   = (const float*)d_in[2];
  const float* watt  = (const float*)d_in[3];
  const float* Wih_f = (const float*)d_in[4];
  const float* Whh_f = (const float*)d_in[5];
  const float* bih_f = (const float*)d_in[6];
  const float* bhh_f = (const float*)d_in[7];
  const float* Wih_b = (const float*)d_in[8];
  const float* Whh_b = (const float*)d_in[9];
  const float* bih_b = (const float*)d_in[10];
  const float* bhh_b = (const float*)d_in[11];
  const float* p1w   = (const float*)d_in[12];
  const float* p1b   = (const float*)d_in[13];
  float* out = (float*)d_out;

  char* ws = (char*)d_ws;
  size_t off = 0;
  auto alloc = [&](size_t bytes) -> char* {
    char* p = ws + off;
    off += (bytes + 255) & ~(size_t)255;
    return p;
  };

  // --- zero-initialized region first (one zero kernel covers it all) ---
  float*          sumMf = (float*)alloc((size_t)NL * D * 4);
  float*          sumMb = (float*)alloc((size_t)NL * D * 4);
  float*          hf    = (float*)alloc((size_t)T * D * 4);
  float*          hb    = (float*)alloc((size_t)T * D * 4);
  unsigned short* hf_bf = (unsigned short*)alloc((size_t)T * D * 2);
  unsigned short* hb_bf = (unsigned short*)alloc((size_t)T * D * 2);
  const size_t zero_bytes = off;

  // --- remaining workspace ---
  float*          ec    = (float*)alloc((size_t)BS * T * D * 4);
  unsigned short* ecw3  = (unsigned short*)alloc((size_t)BS * T * D * 2);
  unsigned short* eqb   = (unsigned short*)alloc((size_t)BS * J * D * 2);
  unsigned short* eqT   = (unsigned short*)alloc((size_t)BS * D * J * 2);
  float*          s2v   = (float*)alloc((size_t)BS * T * J * 4);
  unsigned short* att   = (unsigned short*)alloc((size_t)BS * T * J * 2);
  float*          c2q   = (float*)alloc((size_t)BS * T * D * 4);
  unsigned short* Gbf   = (unsigned short*)alloc((size_t)BS * T * G3 * 2);
  float*          xWf   = (float*)alloc((size_t)BS * T * G3 * 4);
  float*          xWb   = (float*)alloc((size_t)BS * T * G3 * 4);
  float*          hW    = (float*)alloc((size_t)T * G3 * 4);
  float*          sumG  = (float*)alloc((size_t)BS * G3 * 4);
  unsigned short* GMbf  = (unsigned short*)alloc((size_t)BS * D5 * 2);
  float*          logit = (float*)alloc((size_t)BS * T * 4);
  unsigned short* Wihf_bf = (unsigned short*)alloc((size_t)G3 * G3 * 2);
  unsigned short* Wihb_bf = (unsigned short*)alloc((size_t)G3 * G3 * 2);
  unsigned short* Whhf_bf = (unsigned short*)alloc((size_t)G3 * D * 2);
  unsigned short* Whhb_bf = (unsigned short*)alloc((size_t)G3 * D * 2);
  unsigned short* p1w_bf  = (unsigned short*)alloc((size_t)T * D5 * 2);

  // --- init + weight conversion ---
  zero_ws<<<512, 256, 0, stream>>>((float*)ws, (long)(zero_bytes / 4));
  cvt_bf<<<2304, 256, 0, stream>>>(Wih_f, Wihf_bf, (long)G3 * G3);
  cvt_bf<<<2304, 256, 0, stream>>>(Wih_b, Wihb_bf, (long)G3 * G3);
  cvt_bf<<<768,  256, 0, stream>>>(Whh_f, Whhf_bf, (long)G3 * D);
  cvt_bf<<<768,  256, 0, stream>>>(Whh_b, Whhb_bf, (long)G3 * D);
  cvt_bf<<<5120, 256, 0, stream>>>(p1w,   p1w_bf,  (long)T * D5);

  // --- embedding gathers ---
  gather_ctx<<<BS * T, 256, 0, stream>>>(ctx, emb, watt, ec, ecw3);
  gather_q<<<BS * J, 256, 0, stream>>>(qry, emb, eqb, eqT);

  // --- attention: s2 = (ec*w3) @ eq^T ; softmax ; c2q = A @ eq ---
  gemm_bf16<<<dim3(T / BM, 1, BS), 256, 0, stream>>>(
      ecw3, eqb, s2v, nullptr, T, J, D, (long)T * D, (long)J * D, (long)T * J);
  softmax_att<<<BS * T / 4, 256, 0, stream>>>(s2v, att);
  gemm_bf16<<<dim3(T / BM, D / BN, BS), 256, 0, stream>>>(
      att, eqT, c2q, nullptr, T, D, J, (long)T * J, (long)D * J, (long)T * D);

  // --- G, input projections, G-sum ---
  build_g<<<BS * T, 256, 0, stream>>>(ec, c2q, Gbf);
  gemm_bf16<<<dim3(BS * T / BM, G3 / BN, 1), 256, 0, stream>>>(
      Gbf, Wihf_bf, xWf, bih_f, BS * T, G3, G3, 0, 0, 0);
  gemm_bf16<<<dim3(BS * T / BM, G3 / BN, 1), 256, 0, stream>>>(
      Gbf, Wihb_bf, xWb, bih_b, BS * T, G3, G3, 0, 0, 0);
  sum_g<<<BS * 12, 256, 0, stream>>>(ec, c2q, sumG);

  // --- GRU scans (32 steps; batch = 1024 independent rows) ---
  for (int l = 0; l < NL; ++l) {
    gemm_bf16<<<dim3(T / BM, G3 / BN, 1), 256, 0, stream>>>(
        hf_bf, Whhf_bf, hW, bhh_f, T, G3, D, 0, 0, 0);
    gru_gate<<<16, 256, 0, stream>>>(xWf + (size_t)l * T * G3, hW, hf, hf_bf, sumMf + (size_t)l * D);
  }
  for (int l = NL - 1; l >= 0; --l) {
    gemm_bf16<<<dim3(T / BM, G3 / BN, 1), 256, 0, stream>>>(
        hb_bf, Whhb_bf, hW, bhh_b, T, G3, D, 0, 0, 0);
    gru_gate<<<16, 256, 0, stream>>>(xWb + (size_t)l * T * G3, hW, hb, hb_bf, sumMb + (size_t)l * D);
  }

  // --- head: GM -> logits -> log_softmax ---
  build_gm<<<BS, 256, 0, stream>>>(sumG, sumMf, sumMb, GMbf);
  gemm_bf16<<<dim3(1, T / BN, 1), 256, 0, stream>>>(
      GMbf, p1w_bf, logit, p1b, BS, T, D5, 0, 0, 0);
  logsoftmax_rows<<<BS, 256, 0, stream>>>(logit, out);
}
